// MyModel_61933428410970
// MI455X (gfx1250) — matrix operand order repaired and checked by reading, not compiled, not run on hardware
//
#include <hip/hip_runtime.h>
#include <cmath>

// ============================================================================
// Single-head attention (B=8, S=2048, D=768) on gfx1250.
// All matmuls: v_wmma_f32_16x16x32_bf16; tile staging via Tensor Data Mover
// (tensor_load_to_lds + s_wait_tensorcnt) with double-buffered LDS.
//
// Every GEMM B operand is kept in [N][K] (k-contiguous) layout so both A and B
// fragments are plain 2x b128 LDS loads (no gather/pack VALU).
//
// Pipeline (intermediates bf16 in d_ws, ~173 MB, L2-resident):
//   0) w_qkv, w_out -> bf16 TRANSPOSED copies  [N][K]
//   1) qkv = x @ w_qkvT + b_qkv -> q[t][d], k[t][d], vT[b][d][s]
//   2) S = (q @ k^T) * 1/sqrt(D)   (B = k[t][d] already [N][K])
//   3) row softmax over S (in place)
//   4) o = P @ v                   (B = vT)
//   5) out = o @ w_outT + b_out -> fp32 d_out
// ============================================================================

typedef __attribute__((ext_vector_type(16))) __bf16 v16bf;
typedef __attribute__((ext_vector_type(8)))  float  v8f;

#define D_MODEL 768
#define SEQ     2048
#define BATCH   8
#define M_TOK   (BATCH * SEQ)
static const size_t QKV_SZ = (size_t)M_TOK * D_MODEL;

__device__ __forceinline__ unsigned short f2bf(float f) {
  unsigned u = __float_as_uint(f);
  u += 0x7FFFu + ((u >> 16) & 1u);
  return (unsigned short)(u >> 16);
}
__device__ __forceinline__ unsigned pack2(float lo, float hi) {
  return (unsigned)f2bf(lo) | ((unsigned)f2bf(hi) << 16);
}
__device__ __forceinline__ float bf2f(unsigned short h) {
  return __uint_as_float(((unsigned)h) << 16);
}

// ---------------------------------------------------------------------------
// Tensor Data Mover: 2D bf16 tile (tile_d0 x tile_d1, k-contiguous) -> LDS.
// Descriptor bit layout per CDNA5 ISA ch.8 (D# groups 0/1; groups 2/3 zero).
// ---------------------------------------------------------------------------
#ifdef __has_builtin
#if __has_builtin(__builtin_amdgcn_tensor_load_to_lds) && \
    __has_builtin(__builtin_amdgcn_s_wait_tensorcnt)
#define HAVE_TDM 1
#endif
#endif
#ifndef HAVE_TDM
#define HAVE_TDM 0
#endif

#if __has_include(<hip/amd_detail/amd_gfx1250_TDM.h>)
#define TDM_SIX_ARGS 1   // amdgpu-toolchain / therock headers -> 6-arg builtin
#else
#define TDM_SIX_ARGS 0   // ROCm 7.2 -> 5-arg builtin
#endif

#if HAVE_TDM
typedef __attribute__((ext_vector_type(4))) unsigned tdm_v4u;
typedef __attribute__((ext_vector_type(8))) int      tdm_v8i;
typedef __attribute__((ext_vector_type(4))) int      tdm_v4i;

__device__ __forceinline__ void tdm_load_2d(const void* lds_dst,
                                            const void* gsrc,
                                            unsigned tensor_d0,   // row length (elems)
                                            unsigned tensor_d1,   // #rows
                                            unsigned tile_d0,     // tile width (elems)
                                            unsigned tile_d1,     // tile #rows
                                            unsigned long long stride0) // elems
{
  unsigned long long ga = (unsigned long long)(size_t)gsrc;
  unsigned lds_addr = (unsigned)(unsigned long long)(size_t)lds_dst; // addr[31:0] = LDS byte offset
  tdm_v4u g0;
  g0[0] = 1u;                                   // count=1, user mode, no gather
  g0[1] = lds_addr;                             // lds_addr[31:0]
  g0[2] = (unsigned)ga;                         // global_addr[31:0]
  g0[3] = (unsigned)(ga >> 32) | (2u << 30);    // global_addr[56:32] | type=2
  tdm_v8i g1;
  g1[0] = 0x00010000;                           // wg_mask=0, data_size=1 (2B)
  g1[1] = (int)((tensor_d0 & 0xFFFFu) << 16);   // tensor_dim0[15:0]
  g1[2] = (int)((tensor_d0 >> 16) | ((tensor_d1 & 0xFFFFu) << 16));
  g1[3] = (int)((tensor_d1 >> 16) | (tile_d0 << 16));
  g1[4] = (int)tile_d1;                         // tile_dim1; tile_dim2=0
  g1[5] = (int)(unsigned)stride0;               // tensor_dim0_stride[31:0]
  g1[6] = (int)(unsigned)(stride0 >> 32);       // stride[47:32]; dim1_stride lo16=0
  g1[7] = 0;
  tdm_v4i z4 = {0, 0, 0, 0};
#if TDM_SIX_ARGS
  tdm_v8i z8 = {0, 0, 0, 0, 0, 0, 0, 0};
  __builtin_amdgcn_tensor_load_to_lds(g0, g1, z4, z4, z8, 0);
#else
  __builtin_amdgcn_tensor_load_to_lds(g0, g1, z4, z4, 0);
#endif
}
#endif // HAVE_TDM

// ----------------------------------------------------------------------------
// WMMA GEMM: C[M,N] = scale*(A[M,K] @ Bt[N,K]^T) (+ bias). 128x128x32 tiles,
// 256 thr = 8 waves (wave tile 32x64), double-buffered LDS, TDM staging.
// OUT_MODE: 0 fp32, 1 bf16, 2 bf16 split q/k/vT
// ----------------------------------------------------------------------------
template<bool A_BF16, int OUT_MODE, bool ADD_BIAS>
__global__ __launch_bounds__(256)
void gemm_bf16_wmma(const float* __restrict__ Af,
                    const unsigned short* __restrict__ Ab, int lda, int Mrows,
                    const unsigned short* __restrict__ Bt, int ldb, int Ncols,
                    float* __restrict__ Cf, unsigned short* __restrict__ Cb, int ldc,
                    const float* __restrict__ bias, float scale, int K,
                    size_t a_bs, size_t b_bs, size_t c_bs)
{
  __shared__ unsigned short sA[2][128 * 32];   // [m][k] tile, k-contig (2x8KB)
  __shared__ unsigned short sB[2][128 * 32];   // [n][k] tile, k-contig (2x8KB)

  const int t    = threadIdx.x;
  const int lane = t & 31;
  const int wave = t >> 5;
  const int half = lane >> 4;
  const int lr   = lane & 15;
  const int wrow = (wave & 3) * 32;
  const int wcol = (wave >> 2) * 64;

  const int n0 = blockIdx.x * 128;
  const int m0 = blockIdx.y * 128;
  const size_t z = blockIdx.z;

  const float*          AFz = A_BF16 ? nullptr : (Af + z * a_bs);
  const unsigned short* ABz = A_BF16 ? (Ab + z * a_bs) : nullptr;
  const unsigned short* Bz  = Bt + z * b_bs;
  const int nk = K / 32;

  // thread -> staging coords
  const int srow = t >> 1;              // 0..127
  const int scol = (t & 1) * 16;        // 0 or 16

  v8f acc[2][4];
#pragma unroll
  for (int im = 0; im < 2; ++im)
#pragma unroll
    for (int in = 0; in < 4; ++in)
#pragma unroll
      for (int e = 0; e < 8; ++e) acc[im][in][e] = 0.0f;

  // ---------------- prologue: stage tile 0 into buffer 0 -------------------
  if (!A_BF16) {
    const float4* ap = reinterpret_cast<const float4*>(
        AFz + (size_t)(m0 + srow) * lda + scol);
    float4 f0 = ap[0], f1 = ap[1], f2 = ap[2], f3 = ap[3];
    unsigned* sp = reinterpret_cast<unsigned*>(&sA[0][srow * 32 + scol]);
    sp[0] = pack2(f0.x, f0.y); sp[1] = pack2(f0.z, f0.w);
    sp[2] = pack2(f1.x, f1.y); sp[3] = pack2(f1.z, f1.w);
    sp[4] = pack2(f2.x, f2.y); sp[5] = pack2(f2.z, f2.w);
    sp[6] = pack2(f3.x, f3.y); sp[7] = pack2(f3.z, f3.w);
  }
#if HAVE_TDM
  if (A_BF16 && wave == 0)
    tdm_load_2d(&sA[0][0], ABz + (size_t)m0 * lda,
                (unsigned)lda, (unsigned)Mrows, 32u, 128u,
                (unsigned long long)lda);
  if (wave == 1)
    tdm_load_2d(&sB[0][0], Bz + (size_t)n0 * ldb,
                (unsigned)ldb, (unsigned)Ncols, 32u, 128u,
                (unsigned long long)ldb);
  if (wave < 2) __builtin_amdgcn_s_wait_tensorcnt(0);
#else
  if (A_BF16) {
    const uint4* ap = reinterpret_cast<const uint4*>(
        ABz + (size_t)(m0 + srow) * lda + scol);
    uint4 w0 = ap[0], w1 = ap[1];
    uint4* sp = reinterpret_cast<uint4*>(&sA[0][srow * 32 + scol]);
    sp[0] = w0; sp[1] = w1;
  }
  {
    const uint4* bp = reinterpret_cast<const uint4*>(
        Bz + (size_t)(n0 + srow) * ldb + scol);
    uint4 w0 = bp[0], w1 = bp[1];
    uint4* sp = reinterpret_cast<uint4*>(&sB[0][srow * 32 + scol]);
    sp[0] = w0; sp[1] = w1;
  }
#endif
  __syncthreads();

  // ---------------- main loop ----------------------------------------------
  for (int kt = 0; kt < nk; ++kt) {
    const int cur = kt & 1;
    const int nxt = cur ^ 1;
    const bool hn = (kt + 1) < nk;
    const int k1 = (kt + 1) * 32;

#if HAVE_TDM
    // issue DMA for next tiles; overlaps with this tile's WMMAs
    if (hn) {
      if (A_BF16 && wave == 0)
        tdm_load_2d(&sA[nxt][0], ABz + (size_t)m0 * lda + k1,
                    (unsigned)lda, (unsigned)Mrows, 32u, 128u,
                    (unsigned long long)lda);
      if (wave == 1)
        tdm_load_2d(&sB[nxt][0], Bz + (size_t)n0 * ldb + k1,
                    (unsigned)ldb, (unsigned)Ncols, 32u, 128u,
                    (unsigned long long)ldb);
    }
#endif
    // register prefetch for manually-staged operands (overlaps compute)
    float4 f0, f1, f2, f3;
#if !HAVE_TDM
    uint4 wa0, wa1, wb0, wb1;
#endif
    if (hn) {
      if (!A_BF16) {
        const float4* ap = reinterpret_cast<const float4*>(
            AFz + (size_t)(m0 + srow) * lda + k1 + scol);
        f0 = ap[0]; f1 = ap[1]; f2 = ap[2]; f3 = ap[3];
      }
#if !HAVE_TDM
      else {
        const uint4* ap = reinterpret_cast<const uint4*>(
            ABz + (size_t)(m0 + srow) * lda + k1 + scol);
        wa0 = ap[0]; wa1 = ap[1];
      }
      {
        const uint4* bp = reinterpret_cast<const uint4*>(
            Bz + (size_t)(n0 + srow) * ldb + k1 + scol);
        wb0 = bp[0]; wb1 = bp[1];
      }
#endif
    }

    // ---- fragments: all 2x b128 LDS loads (k-contiguous layouts) ----
    union Frag { v16bf v; uint4 q[2]; };
    Frag a[2], b[4];
#pragma unroll
    for (int im = 0; im < 2; ++im) {
      // A 16x32: lanes 0-15 K=0..7|16..23, lanes 16-31 K=8..15|24..31
      const unsigned short* ar = &sA[cur][(wrow + im * 16 + lr) * 32];
      a[im].q[0] = *reinterpret_cast<const uint4*>(&ar[8 * half]);
      a[im].q[1] = *reinterpret_cast<const uint4*>(&ar[16 + 8 * half]);
    }
#pragma unroll
    for (int in = 0; in < 4; ++in) {
      // B 32x16: lanes 0-15 K=0..7|16..23, lanes 16-31 K=8..15|24..31 (col = lr)
      const unsigned short* br = &sB[cur][(wcol + in * 16 + lr) * 32];
      b[in].q[0] = *reinterpret_cast<const uint4*>(&br[8 * half]);
      b[in].q[1] = *reinterpret_cast<const uint4*>(&br[16 + 8 * half]);
    }
#pragma unroll
    for (int im = 0; im < 2; ++im)
#pragma unroll
      for (int in = 0; in < 4; ++in)
        acc[im][in] = __builtin_amdgcn_wmma_f32_16x16x32_bf16(
            false, a[im].v, false, b[in].v, (short)0, acc[im][in], false, false);

    // ---- write prefetched data into the other buffer ----
    if (hn) {
      if (!A_BF16) {
        unsigned* sp = reinterpret_cast<unsigned*>(&sA[nxt][srow * 32 + scol]);
        sp[0] = pack2(f0.x, f0.y); sp[1] = pack2(f0.z, f0.w);
        sp[2] = pack2(f1.x, f1.y); sp[3] = pack2(f1.z, f1.w);
        sp[4] = pack2(f2.x, f2.y); sp[5] = pack2(f2.z, f2.w);
        sp[6] = pack2(f3.x, f3.y); sp[7] = pack2(f3.z, f3.w);
      }
#if !HAVE_TDM
      else {
        uint4* sp = reinterpret_cast<uint4*>(&sA[nxt][srow * 32 + scol]);
        sp[0] = wa0; sp[1] = wa1;
      }
      {
        uint4* sp = reinterpret_cast<uint4*>(&sB[nxt][srow * 32 + scol]);
        sp[0] = wb0; sp[1] = wb1;
      }
#endif
    }
#if HAVE_TDM
    if (hn && wave < 2) __builtin_amdgcn_s_wait_tensorcnt(0);
#endif
    __syncthreads();
  }

  // ---------------- epilogue ------------------------------------------------
  float*          Cfz = (OUT_MODE == 0) ? (Cf + z * c_bs) : nullptr;
  unsigned short* Cbz = (OUT_MODE != 0) ? (Cb + z * c_bs) : nullptr;
#pragma unroll
  for (int im = 0; im < 2; ++im) {
#pragma unroll
    for (int in = 0; in < 4; ++in) {
      const int gcol = n0 + wcol + in * 16 + lr;
      const float bv = ADD_BIAS ? bias[gcol] : 0.0f;
#pragma unroll
      for (int j = 0; j < 8; ++j) {
        const int grow = m0 + wrow + im * 16 + half * 8 + j;
        const float v = acc[im][in][j] * scale + bv;
        if (OUT_MODE == 0) {
          Cfz[(size_t)grow * ldc + gcol] = v;
        } else if (OUT_MODE == 1) {
          Cbz[(size_t)grow * ldc + gcol] = f2bf(v);
        } else {
          if (gcol < D_MODEL) {
            Cbz[(size_t)grow * D_MODEL + gcol] = f2bf(v);                 // q[t][d]
          } else if (gcol < 2 * D_MODEL) {
            (Cbz + QKV_SZ)[(size_t)grow * D_MODEL + (gcol - D_MODEL)] = f2bf(v); // k[t][d]
          } else {
            const int bb = grow >> 11, sr = grow & (SEQ - 1);             // vT[b][d][s]
            (Cbz + 2 * QKV_SZ)[(size_t)bb * D_MODEL * SEQ +
                               (size_t)(gcol - 2 * D_MODEL) * SEQ + sr] = f2bf(v);
          }
        }
      }
    }
  }
}

// ----------------------------------------------------------------------------
__global__ __launch_bounds__(256)
void softmax_rows(unsigned short* __restrict__ S)
{
  __shared__ float red[8];
  const int t = threadIdx.x;
  unsigned short* row = S + (size_t)blockIdx.x * SEQ;

  float x[8];
  float lmax = -3.4e38f;
#pragma unroll
  for (int i = 0; i < 8; ++i) {
    x[i] = bf2f(row[t + i * 256]);
    lmax = fmaxf(lmax, x[i]);
  }
#pragma unroll
  for (int off = 16; off > 0; off >>= 1)
    lmax = fmaxf(lmax, __shfl_xor(lmax, off, 32));
  if ((t & 31) == 0) red[t >> 5] = lmax;
  __syncthreads();
  float rmax = red[0];
#pragma unroll
  for (int i = 1; i < 8; ++i) rmax = fmaxf(rmax, red[i]);
  __syncthreads();

  float lsum = 0.0f;
#pragma unroll
  for (int i = 0; i < 8; ++i) { x[i] = __expf(x[i] - rmax); lsum += x[i]; }
#pragma unroll
  for (int off = 16; off > 0; off >>= 1)
    lsum += __shfl_xor(lsum, off, 32);
  if ((t & 31) == 0) red[t >> 5] = lsum;
  __syncthreads();
  float rsum = 0.0f;
#pragma unroll
  for (int i = 0; i < 8; ++i) rsum += red[i];
  const float inv = 1.0f / rsum;
#pragma unroll
  for (int i = 0; i < 8; ++i) row[t + i * 256] = f2bf(x[i] * inv);
}

// fp32 [K][N] -> bf16 transposed [N][K]
__global__ void f32_to_bf16_T(const float* __restrict__ src,
                              unsigned short* __restrict__ dst,
                              int rows, int cols)
{
  int i = blockIdx.x * 256 + threadIdx.x;
  if (i < rows * cols) {
    int n = i / rows, k = i - n * rows;
    dst[i] = f2bf(src[(size_t)k * cols + n]);
  }
}

// ----------------------------------------------------------------------------
extern "C" void kernel_launch(void* const* d_in, const int* in_sizes, int n_in,
                              void* d_out, int out_size, void* d_ws, size_t ws_size,
                              hipStream_t stream)
{
  (void)in_sizes; (void)n_in; (void)out_size; (void)ws_size;
  const float* x     = (const float*)d_in[0];
  const float* w_qkv = (const float*)d_in[1];
  const float* b_qkv = (const float*)d_in[2];
  const float* w_out = (const float*)d_in[3];
  const float* b_out = (const float*)d_in[4];
  float* out = (float*)d_out;

  // workspace (bf16 elems), ~173 MB total, L2-resident
  unsigned short* ws     = (unsigned short*)d_ws;
  unsigned short* qb     = ws;                                  // [16384][768]
  unsigned short* kb     = qb + QKV_SZ;                         // [16384][768]
  unsigned short* vT     = kb + QKV_SZ;                         // [8][768][2048]
  unsigned short* Sb     = vT + QKV_SZ;                         // [8][2048][2048]
  unsigned short* ob     = Sb + (size_t)BATCH * SEQ * SEQ;      // [16384][768]
  unsigned short* wqkvT  = ob + QKV_SZ;                         // [2304][768]
  unsigned short* woutT  = wqkvT + (size_t)D_MODEL * 3 * D_MODEL; // [768][768]

  const int NWQKV = D_MODEL * 3 * D_MODEL;
  const int NWOUT = D_MODEL * D_MODEL;
  f32_to_bf16_T<<<(NWQKV + 255) / 256, 256, 0, stream>>>(w_qkv, wqkvT, D_MODEL, 3 * D_MODEL);
  f32_to_bf16_T<<<(NWOUT + 255) / 256, 256, 0, stream>>>(w_out, woutT, D_MODEL, D_MODEL);

  // 1) QKV projection (A fp32 converted in-flight), split -> q, k, vT
  gemm_bf16_wmma<false, 2, true>
      <<<dim3(3 * D_MODEL / 128, M_TOK / 128, 1), 256, 0, stream>>>(
      x, nullptr, D_MODEL, M_TOK, wqkvT, D_MODEL, 3 * D_MODEL,
      nullptr, qb, D_MODEL, b_qkv, 1.0f, D_MODEL, 0, 0, 0);

  // 2) S = (q @ k^T) * 1/sqrt(D)
  const float scale = 1.0f / sqrtf((float)D_MODEL);
  gemm_bf16_wmma<true, 1, false>
      <<<dim3(SEQ / 128, SEQ / 128, BATCH), 256, 0, stream>>>(
      nullptr, qb, D_MODEL, SEQ, kb, D_MODEL, SEQ,
      nullptr, Sb, SEQ, nullptr, scale, D_MODEL,
      (size_t)SEQ * D_MODEL, (size_t)SEQ * D_MODEL, (size_t)SEQ * SEQ);

  // 3) row softmax
  softmax_rows<<<dim3(BATCH * SEQ), 256, 0, stream>>>(Sb);

  // 4) o = P @ v   (B = vT[b][d][s])
  gemm_bf16_wmma<true, 1, false>
      <<<dim3(D_MODEL / 128, SEQ / 128, BATCH), 256, 0, stream>>>(
      nullptr, Sb, SEQ, SEQ, vT, SEQ, D_MODEL,
      nullptr, ob, D_MODEL, nullptr, 1.0f, SEQ,
      (size_t)SEQ * SEQ, (size_t)D_MODEL * SEQ, (size_t)SEQ * D_MODEL);

  // 5) out = o @ w_outT + b_out -> fp32
  gemm_bf16_wmma<true, 0, true>
      <<<dim3(D_MODEL / 128, M_TOK / 128, 1), 256, 0, stream>>>(
      nullptr, ob, D_MODEL, M_TOK, woutT, D_MODEL, D_MODEL,
      out, nullptr, D_MODEL, b_out, 1.0f, D_MODEL, 0, 0, 0);
}
